// VQGAN_76931454206655
// MI455X (gfx1250) — matrix operand order repaired and checked by reading, not compiled, not run on hardware
//
#include <hip/hip_runtime.h>
#include <hip/hip_bf16.h>

// ---------------------------------------------------------------------------
// CDNA5 (gfx1250) VQGAN forward: all convs via v_wmma_f32_16x16x32_f16
// implicit GEMM; ConvTranspose k4s2p1 decomposed into 4 phase 2x2 convs.
// GEMM: 16x64 tile per wave, ping/pong K-chunks (K padded to 64) so the
// global_load_b128s of one chunk overlap the 4 WMMAs of the other chunk
// without register copies.
// ---------------------------------------------------------------------------

typedef _Float16 v16h __attribute__((ext_vector_type(16)));
typedef _Float16 v8h  __attribute__((ext_vector_type(8)));
typedef float    v8f  __attribute__((ext_vector_type(8)));

static inline int pad16i(int x) { return (x + 15) & ~15; }
static inline int pad64i(int x) { return (x + 63) & ~63; }

#define WMMA_F16(va, vb, c) \
  __builtin_amdgcn_wmma_f32_16x16x32_f16(false, (va), false, (vb), (short)0, (c), false, false)

// ---------------------------------------------------------------------------
// Weight packing: OIHW fp32 -> fp16 A-matrix [Mp][Kp], zero padded.
// ---------------------------------------------------------------------------
__global__ void pack_conv_w(const float* __restrict__ w, _Float16* __restrict__ A,
                            int Co, int Ci, int KH, int KW, int Mp, int Kp) {
  int idx = blockIdx.x * 256 + threadIdx.x;
  int tot = Mp * Kp;
  if (idx >= tot) return;
  int m = idx / Kp, k = idx % Kp;
  int K = Ci * KH * KW;
  float v = 0.0f;
  if (m < Co && k < K) v = w[m * K + k];
  A[idx] = (_Float16)v;
}

// ConvTranspose2d weight (Ci,Co,4,4); phase (py,px) uses 2x2 taps:
//   py==0: a=0 -> ky=1 (dy=0),  a=1 -> ky=3 (dy=-1)
//   py==1: a=0 -> ky=0 (dy=+1), a=1 -> ky=2 (dy=0)      (same for x)
__global__ void pack_deconv_w(const float* __restrict__ w, _Float16* __restrict__ A,
                              int Ci, int Co, int py, int px, int Mp, int Kp) {
  int idx = blockIdx.x * 256 + threadIdx.x;
  int tot = Mp * Kp;
  if (idx >= tot) return;
  int m = idx / Kp, k = idx % Kp;
  float v = 0.0f;
  if (m < Co && k < Ci * 4) {
    int ci = k >> 2, a = (k >> 1) & 1, bb = k & 1;
    int ky = py ? (a ? 2 : 0) : (a ? 3 : 1);
    int kx = px ? (bb ? 2 : 0) : (bb ? 3 : 1);
    v = w[((ci * Co + m) * 4 + ky) * 4 + kx];
  }
  A[idx] = (_Float16)v;
}

// ---------------------------------------------------------------------------
// im2col for one batch image: NCHW fp32 -> Bt[n][k] fp16 (k contiguous per
// output position n, so each lane's WMMA B fragment is 16 contiguous halves).
// ---------------------------------------------------------------------------
__global__ void im2col_conv(const float* __restrict__ x, _Float16* __restrict__ Bt,
                            int b, int Ci, int H, int W, int KH, int KW,
                            int stride, int pad, int OH, int OW, int Kp) {
  unsigned idx = blockIdx.x * 256 + threadIdx.x;
  unsigned tot = (unsigned)(OH * OW) * (unsigned)Kp;
  if (idx >= tot) return;
  unsigned n = idx / (unsigned)Kp, k = idx % (unsigned)Kp;
  float v = 0.0f;
  unsigned K = (unsigned)(Ci * KH * KW);
  if (k < K) {
    unsigned kk = (unsigned)(KH * KW);
    unsigned ci = k / kk, r = k % kk, kh = r / (unsigned)KW, kw = r % (unsigned)KW;
    int oy = (int)(n / (unsigned)OW), ox = (int)(n % (unsigned)OW);
    int iy = oy * stride - pad + (int)kh;
    int ix = ox * stride - pad + (int)kw;
    if (iy >= 0 && iy < H && ix >= 0 && ix < W)
      v = x[(((long)b * Ci + ci) * H + iy) * W + ix];
  }
  Bt[(long)n * Kp + k] = (_Float16)v;
}

// im2col for one deconv phase: patch = 2x2 taps around (ty,tx) on input grid.
__global__ void im2col_deconv(const float* __restrict__ x, _Float16* __restrict__ Bt,
                              int b, int Ci, int H, int W, int py, int px, int Kp) {
  unsigned idx = blockIdx.x * 256 + threadIdx.x;
  unsigned tot = (unsigned)(H * W) * (unsigned)Kp;
  if (idx >= tot) return;
  unsigned n = idx / (unsigned)Kp, k = idx % (unsigned)Kp;
  float v = 0.0f;
  if (k < (unsigned)(Ci * 4)) {
    int ci = (int)(k >> 2), a = (int)((k >> 1) & 1), bb = (int)(k & 1);
    int ty = (int)(n / (unsigned)W), tx = (int)(n % (unsigned)W);
    int dy = py ? (a ? 0 : 1) : (a ? -1 : 0);
    int dx = px ? (bb ? 0 : 1) : (bb ? -1 : 0);
    int iy = ty + dy, ix = tx + dx;
    if (iy >= 0 && iy < H && ix >= 0 && ix < W)
      v = x[(((long)b * Ci + ci) * H + iy) * W + ix];
  }
  Bt[(long)n * Kp + k] = (_Float16)v;
}

// ---------------------------------------------------------------------------
// WMMA GEMM: 16(M) x 64(N) tile per wave, 4 accumulators, A fragment shared
// across the 4 WMMAs. Kp is a multiple of 64; the K loop is 2x unrolled with
// ping(P)/pong(Q) fragment sets: Q's loads are issued before P's WMMAs (and
// vice versa), so vmem latency overlaps matrix work with no register copies.
// ---------------------------------------------------------------------------
__global__ __launch_bounds__(128)
void gemm_wmma(const _Float16* __restrict__ A, const _Float16* __restrict__ Bt,
               const float* __restrict__ bias, const float* __restrict__ res,
               float* __restrict__ out,
               int Mp, int Kp, int M, int b, int gwShift,
               int ystep, int yoff, int xstep, int xoff,
               int outH, int outW, int relu) {
  int lane = threadIdx.x & 31;
  int wave = threadIdx.x >> 5;
  int n0 = (blockIdx.x * 4 + wave) * 64;   // 4 waves/block, 64 columns/wave
  int m0 = blockIdx.y * 16;
  int hi  = lane >> 4;     // 0: lanes 0-15, 1: lanes 16-31
  int l15 = lane & 15;

  // A: lane holds row m0+l15; K elems k0..k0+15 (+16 for hi half)
  const _Float16* arow = A  + (long)(m0 + l15) * Kp + hi * 16;
  // B: lane holds column; K elems k0..k0+15 (+16 for hi half)
  const _Float16* bc   = Bt + (long)(n0 + l15) * Kp + hi * 16;
  const long cs = (long)16 * Kp;           // column-subtile stride (16 columns)

  v8f acc[4] = {{}, {}, {}, {}};

  // preload K-chunk 0 into fragment set P
  v8h  pal = *(const v8h*)(arow);
  v8h  pah = *(const v8h*)(arow + 8);
  v16h pb0 = *(const v16h*)(bc);
  v16h pb1 = *(const v16h*)(bc + cs);
  v16h pb2 = *(const v16h*)(bc + 2 * cs);
  v16h pb3 = *(const v16h*)(bc + 3 * cs);

  for (int k0 = 0; k0 < Kp; k0 += 64) {
    // issue loads for chunk k0+32 into fragment set Q
    const _Float16* aq = arow + k0 + 32;
    const _Float16* bq = bc + k0 + 32;
    v8h  qal = *(const v8h*)(aq);
    v8h  qah = *(const v8h*)(aq + 8);
    v16h qb0 = *(const v16h*)(bq);
    v16h qb1 = *(const v16h*)(bq + cs);
    v16h qb2 = *(const v16h*)(bq + 2 * cs);
    v16h qb3 = *(const v16h*)(bq + 3 * cs);

    __builtin_prefetch((const void*)(arow + k0 + 64), 0, 1);     // global_prefetch_b8
    __builtin_prefetch((const void*)(bc + k0 + 64), 0, 1);
    __builtin_prefetch((const void*)(bc + cs + k0 + 64), 0, 1);
    __builtin_prefetch((const void*)(bc + 2 * cs + k0 + 64), 0, 1);
    __builtin_prefetch((const void*)(bc + 3 * cs + k0 + 64), 0, 1);

    // WMMAs on P (waits only for P's loads; Q's stay in flight)
    v16h va = __builtin_shufflevector(pal, pah,
                0,1,2,3,4,5,6,7,8,9,10,11,12,13,14,15);
    acc[0] = WMMA_F16(va, pb0, acc[0]);
    acc[1] = WMMA_F16(va, pb1, acc[1]);
    acc[2] = WMMA_F16(va, pb2, acc[2]);
    acc[3] = WMMA_F16(va, pb3, acc[3]);

    // reload P with chunk k0+64 (P is dead here; no copies needed)
    if (k0 + 64 < Kp) {
      const _Float16* ap = arow + k0 + 64;
      const _Float16* bp = bc + k0 + 64;
      pal = *(const v8h*)(ap);
      pah = *(const v8h*)(ap + 8);
      pb0 = *(const v16h*)(bp);
      pb1 = *(const v16h*)(bp + cs);
      pb2 = *(const v16h*)(bp + 2 * cs);
      pb3 = *(const v16h*)(bp + 3 * cs);
    }

    // WMMAs on Q
    v16h vq = __builtin_shufflevector(qal, qah,
                0,1,2,3,4,5,6,7,8,9,10,11,12,13,14,15);
    acc[0] = WMMA_F16(vq, qb0, acc[0]);
    acc[1] = WMMA_F16(vq, qb1, acc[1]);
    acc[2] = WMMA_F16(vq, qb2, acc[2]);
    acc[3] = WMMA_F16(vq, qb3, acc[3]);
  }

  // D layout: VGPR r -> (M = m0 + r + 8*hi, N = subtile_base + l15)
  const int gwMask = (1 << gwShift) - 1;
  #pragma unroll
  for (int s = 0; s < 4; ++s) {
    int n  = n0 + s * 16 + l15;
    int oy = n >> gwShift, ox = n & gwMask;
    int yy = oy * ystep + yoff;
    int xx = ox * xstep + xoff;
    #pragma unroll
    for (int r = 0; r < 8; ++r) {
      int m = m0 + r + hi * 8;
      if (m < M) {
        float v = acc[s][r] + bias[m];
        if (relu) v = v > 0.0f ? v : 0.0f;
        long idx = (((long)b * M + m) * outH + yy) * outW + xx;
        if (res) v += res[idx];
        out[idx] = v;
      }
    }
  }
}

// ---------------------------------------------------------------------------
// Vector quantization: one wave per spatial position. z is NCHW [B,64,H,W];
// forward straight-through == write nearest codebook vector in place.
// ---------------------------------------------------------------------------
__global__ void vq_quantize(float* __restrict__ z, const float* __restrict__ E,
                            int HW) {
  __shared__ float zv[64];
  int lane = threadIdx.x;             // block = 32 (one wave)
  int pos  = blockIdx.x;              // 0 .. B*HW-1
  int bb = pos / HW, s = pos % HW;
  float* zp = z + ((long)bb * 64) * HW + s;
  zv[lane]      = zp[(long)lane * HW];
  zv[lane + 32] = zp[(long)(lane + 32) * HW];
  __syncthreads();

  float best = 3.4e38f;
  int bidx = 0;
  for (int e = lane; e < 512; e += 32) {
    const float* ev = E + e * 64;
    float d = 0.0f;
    #pragma unroll 8
    for (int c = 0; c < 64; ++c) { float t = zv[c] - ev[c]; d += t * t; }
    if (d < best) { best = d; bidx = e; }
  }
  for (int off = 16; off > 0; off >>= 1) {
    float ob = __shfl_xor(best, off, 32);
    int   oi = __shfl_xor(bidx, off, 32);
    if (ob < best || (ob == best && oi < bidx)) { best = ob; bidx = oi; }
  }
  const float* ev = E + bidx * 64;
  zp[(long)lane * HW]        = ev[lane];
  zp[(long)(lane + 32) * HW] = ev[lane + 32];
}

// ---------------------------------------------------------------------------
// Host orchestration
// ---------------------------------------------------------------------------
extern "C" void kernel_launch(void* const* d_in, const int* in_sizes, int n_in,
                              void* d_out, int out_size, void* d_ws, size_t ws_size,
                              hipStream_t stream) {
  (void)in_sizes; (void)n_in; (void)out_size; (void)ws_size;
  const int B = 16;

  auto F = [&](int j) { return (const float*)d_in[j]; };

  // Param leaves in JAX pytree order (dict keys sorted: cb, dec, enc).
  int t = 1;                                      // d_in[0] = x
  const float* E   = F(t++);                      // cb.embed [512,64]
  const float* pjw = F(t++); const float* pjb = F(t++);   // cb.proj (w,b)
  const float *d1w[9], *d1b[9];
  for (int k = 0; k < 9; ++k) { d1w[k] = F(t++); d1b[k] = F(t++); }  // dec.l1
  const float* d2w = F(t++); const float* d2b = F(t++);              // dec.l2
  const float *d3w[9], *d3b[9];
  for (int k = 0; k < 9; ++k) { d3w[k] = F(t++); d3b[k] = F(t++); }  // dec.l3
  const float* d4w = F(t++); const float* d4b = F(t++);              // dec.l4
  const float* dlw = F(t++); const float* dlb = F(t++);              // dec.last
  const float* e1w = F(t++); const float* e1b = F(t++);              // enc.l1
  const float* e2w = F(t++); const float* e2b = F(t++);              // enc.l2
  const float *erw[12], *erb[12];                                    // enc.l3+l4
  for (int k = 0; k < 12; ++k) { erw[k] = F(t++); erb[k] = F(t++); }

  // Workspace layout
  const size_t ACT_S = (size_t)16 * 64 * 128 * 128;   // 16,777,216 floats
  const size_t ACT_L = (size_t)16 * 32 * 256 * 256;   // 33,554,432 floats
  const size_t COL_CAP = (size_t)65536 * 320;         // 20,971,520 halves
  float* B0 = (float*)d_ws;
  float* B1 = B0 + ACT_S;
  float* B2 = B1 + ACT_S;                              // big buffer
  _Float16* colbuf = (_Float16*)(B2 + ACT_L);
  _Float16* wbase  = colbuf + COL_CAP;
  size_t woff = 0;

  auto packConv = [&](const float* w, int Co, int Ci, int KH, int KW) -> _Float16* {
    int Mp = pad16i(Co), Kp = pad64i(Ci * KH * KW);
    _Float16* A = wbase + woff; woff += (size_t)Mp * Kp;
    int tot = Mp * Kp;
    pack_conv_w<<<(tot + 255) / 256, 256, 0, stream>>>(w, A, Co, Ci, KH, KW, Mp, Kp);
    return A;
  };
  auto packDec = [&](const float* w, int Ci, int Co, int py, int px) -> _Float16* {
    int Mp = pad16i(Co), Kp = Ci * 4;                  // 512/256: already /64
    _Float16* A = wbase + woff; woff += (size_t)Mp * Kp;
    int tot = Mp * Kp;
    pack_deconv_w<<<(tot + 255) / 256, 256, 0, stream>>>(w, A, Ci, Co, py, px, Mp, Kp);
    return A;
  };

  auto runConv = [&](const float* in, float* out, const float* res, _Float16* A,
                     const float* bias, int Ci, int H, int W, int Co,
                     int KH, int KW, int st, int pd, int relu) {
    int OH = (H + 2 * pd - KH) / st + 1;
    int OW = (W + 2 * pd - KW) / st + 1;
    int Kp = pad64i(Ci * KH * KW), Mp = pad16i(Co), N = OH * OW;
    int gws = __builtin_ctz(OW);
    for (int b = 0; b < B; ++b) {
      int tot = N * Kp;
      im2col_conv<<<(tot + 255) / 256, 256, 0, stream>>>(
          in, colbuf, b, Ci, H, W, KH, KW, st, pd, OH, OW, Kp);
      gemm_wmma<<<dim3(N / 256, Mp / 16), 128, 0, stream>>>(
          A, colbuf, bias, res, out, Mp, Kp, Co, b, gws, 1, 0, 1, 0, OH, OW, relu);
    }
  };

  auto runDeconv = [&](const float* in, float* out, _Float16* Aph[4],
                       const float* bias, int Ci, int H, int W, int Co) {
    int Kp = Ci * 4, Mp = pad16i(Co), N = H * W;
    int gws = __builtin_ctz(W);
    for (int b = 0; b < B; ++b)
      for (int py = 0; py < 2; ++py)
        for (int px = 0; px < 2; ++px) {
          int tot = N * Kp;
          im2col_deconv<<<(tot + 255) / 256, 256, 0, stream>>>(
              in, colbuf, b, Ci, H, W, py, px, Kp);
          gemm_wmma<<<dim3(N / 256, Mp / 16), 128, 0, stream>>>(
              Aph[py * 2 + px], colbuf, bias, nullptr, out, Mp, Kp, Co, b,
              gws, 2, py, 2, px, 2 * H, 2 * W, /*relu=*/1);
        }
  };

  // ---- pack all weights (deterministic, every call) ----
  _Float16* Ae1 = packConv(e1w, 64, 3, 4, 4);
  _Float16* Ae2 = packConv(e2w, 128, 64, 4, 4);
  _Float16* Aer[12];
  for (int blk = 0; blk < 4; ++blk) {
    Aer[blk * 3 + 0] = packConv(erw[blk * 3 + 0], 64, 128, 1, 1);
    Aer[blk * 3 + 1] = packConv(erw[blk * 3 + 1], 64, 64, 3, 3);
    Aer[blk * 3 + 2] = packConv(erw[blk * 3 + 2], 128, 64, 1, 1);
  }
  _Float16* Apj = packConv(pjw, 64, 128, 1, 1);
  _Float16* Ad1[9];
  Ad1[0] = packConv(d1w[0], 64, 64, 1, 1);     // dec.l1 block0: cin = LAT = 64
  Ad1[1] = packConv(d1w[1], 64, 64, 3, 3);
  Ad1[2] = packConv(d1w[2], 128, 64, 1, 1);
  for (int blk = 1; blk < 3; ++blk) {
    Ad1[blk * 3 + 0] = packConv(d1w[blk * 3 + 0], 64, 128, 1, 1);
    Ad1[blk * 3 + 1] = packConv(d1w[blk * 3 + 1], 64, 64, 3, 3);
    Ad1[blk * 3 + 2] = packConv(d1w[blk * 3 + 2], 128, 64, 1, 1);
  }
  _Float16* Ad2ph[4];
  for (int py = 0; py < 2; ++py)
    for (int px = 0; px < 2; ++px) Ad2ph[py * 2 + px] = packDec(d2w, 128, 64, py, px);
  _Float16* Ad3[9];
  for (int blk = 0; blk < 3; ++blk) {
    Ad3[blk * 3 + 0] = packConv(d3w[blk * 3 + 0], 32, 64, 1, 1);
    Ad3[blk * 3 + 1] = packConv(d3w[blk * 3 + 1], 32, 32, 3, 3);
    Ad3[blk * 3 + 2] = packConv(d3w[blk * 3 + 2], 64, 32, 1, 1);
  }
  _Float16* Ad4ph[4];
  for (int py = 0; py < 2; ++py)
    for (int px = 0; px < 2; ++px) Ad4ph[py * 2 + px] = packDec(d4w, 64, 32, py, px);
  _Float16* Alast = packConv(dlw, 3, 32, 3, 3);

  const float* x = F(0);

  // ---- encoder ----
  runConv(x,  B0, nullptr, Ae1, e1b, 3, 256, 256, 64, 4, 4, 2, 1, 1);   // [64,128,128]
  runConv(B0, B1, nullptr, Ae2, e2b, 64, 128, 128, 128, 4, 4, 2, 1, 1); // [128,64,64]
  {
    const float* X = B1; float* t1 = B2; float* t2 = B0;
    for (int blk = 0; blk < 4; ++blk) {                  // enc.l3 (2) + enc.l4 (2)
      runConv(X,  t1, nullptr, Aer[blk * 3 + 0], erb[blk * 3 + 0], 128, 64, 64, 64, 1, 1, 1, 0, 1);
      runConv(t1, t2, nullptr, Aer[blk * 3 + 1], erb[blk * 3 + 1], 64, 64, 64, 64, 3, 3, 1, 1, 1);
      runConv(t2, t1, X,       Aer[blk * 3 + 2], erb[blk * 3 + 2], 64, 64, 64, 128, 1, 1, 1, 0, 1);
      const float* nx = t1; t1 = t2; t2 = (float*)X; X = nx;
    }
    // encoder output now in B2
  }

  // ---- codebook ----
  runConv(B2, B0, nullptr, Apj, pjb, 128, 64, 64, 64, 1, 1, 1, 0, 0);   // z [64,64,64]
  vq_quantize<<<B * 64 * 64, 32, 0, stream>>>(B0, E, 64 * 64);          // z <- E[argmin]

  // ---- decoder l1: 3 res blocks at 64x64 ----
  {
    const float* X = B0; float* t1 = B1; float* t2 = B2;
    // block0: 64 -> 128, no shortcut
    runConv(X,  t1, nullptr, Ad1[0], d1b[0], 64, 64, 64, 64, 1, 1, 1, 0, 1);
    runConv(t1, t2, nullptr, Ad1[1], d1b[1], 64, 64, 64, 64, 3, 3, 1, 1, 1);
    runConv(t2, t1, nullptr, Ad1[2], d1b[2], 64, 64, 64, 128, 1, 1, 1, 0, 1);
    { const float* nx = t1; t1 = t2; t2 = (float*)X; X = nx; }
    for (int blk = 1; blk < 3; ++blk) {                  // shortcut blocks, 128ch
      runConv(X,  t1, nullptr, Ad1[blk * 3 + 0], d1b[blk * 3 + 0], 128, 64, 64, 64, 1, 1, 1, 0, 1);
      runConv(t1, t2, nullptr, Ad1[blk * 3 + 1], d1b[blk * 3 + 1], 64, 64, 64, 64, 3, 3, 1, 1, 1);
      runConv(t2, t1, X,       Ad1[blk * 3 + 2], d1b[blk * 3 + 2], 64, 64, 64, 128, 1, 1, 1, 0, 1);
      const float* nx = t1; t1 = t2; t2 = (float*)X; X = nx;
    }
    // dec.l1 output in B0
  }

  // ---- deconv l2: [128,64,64] -> relu [64,128,128] ----
  runDeconv(B0, B1, Ad2ph, d2b, 128, 64, 64, 64);

  // ---- decoder l3: 3 res blocks at 128x128, 64ch (inter 32) ----
  {
    const float* X = B1; float* t1 = B2; float* t2 = B0;
    for (int blk = 0; blk < 3; ++blk) {
      runConv(X,  t1, nullptr, Ad3[blk * 3 + 0], d3b[blk * 3 + 0], 64, 128, 128, 32, 1, 1, 1, 0, 1);
      runConv(t1, t2, nullptr, Ad3[blk * 3 + 1], d3b[blk * 3 + 1], 32, 128, 128, 32, 3, 3, 1, 1, 1);
      runConv(t2, t1, X,       Ad3[blk * 3 + 2], d3b[blk * 3 + 2], 32, 128, 128, 64, 1, 1, 1, 0, 1);
      const float* nx = t1; t1 = t2; t2 = (float*)X; X = nx;
    }
    // dec.l3 output in B1
  }

  // ---- deconv l4: [64,128,128] -> relu [32,256,256] ----
  runDeconv(B1, B2, Ad4ph, d4b, 64, 128, 128, 32);

  // ---- last conv: [32,256,256] -> [3,256,256] ----
  runConv(B2, (float*)d_out, nullptr, Alast, dlb, 32, 256, 256, 3, 3, 3, 1, 1, 0);
}
